// EIGNEncoder_63290638074461
// MI455X (gfx1250) — matrix operand order repaired and checked by reading, not compiled, not run on hardware
//
#include <hip/hip_runtime.h>
#include <hip/hip_bf16.h>

// ---------------------------------------------------------------------------
// CDNA5 (gfx1250) EIGN encoder forward.
// Dense linears: v_wmma_f32_16x16x32_bf16, weights staged to LDS with
// global_load_async_to_lds_b128 (ASYNCcnt), one 16 x Dout strip per wave.
// ---------------------------------------------------------------------------

typedef __attribute__((ext_vector_type(16))) __bf16 v16bf;
typedef __attribute__((ext_vector_type(8)))  float  v8f;

#define ACT_NONE 0
#define ACT_SILU 1
#define ACT_BN   2   // leaky_relu(0.01) * BN_SCALE
#define ACT_SIG  3

#define BN_SCALE_F 0.99999500003749985f

__device__ __forceinline__ float apply_act(float v, int act) {
    switch (act) {
        case ACT_SILU: return v / (1.0f + __expf(-v));
        case ACT_BN: { float t = (v > 0.0f) ? v : 0.01f * v; return t * BN_SCALE_F; }
        case ACT_SIG:  return 1.0f / (1.0f + __expf(-v));
        default:       return v;
    }
}

__device__ __forceinline__ void wait_asynccnt0() {
#if __has_builtin(__builtin_amdgcn_s_wait_asynccnt)
    __builtin_amdgcn_s_wait_asynccnt(0);
#else
    asm volatile("s_wait_asynccnt 0x0" ::: "memory");
#endif
}

// Stage Wt[Dout x Din] bf16 (row-major) into LDS with pitch LP = Din+16
// elements, using async global->LDS 16B copies. All 256 threads participate.
__device__ __forceinline__ void stage_weights_async(
    const __bf16* __restrict__ Wt, __bf16* sW, int Din, int Dout, int LP)
{
    unsigned ldsbase = (unsigned)(size_t)sW;   // low 32 bits = LDS byte offset
    int chunks = (Dout * Din) >> 3;            // 8 bf16 = 16 bytes per chunk
    for (int c = threadIdx.x; c < chunks; c += 256) {
        int el  = c << 3;
        int row = el / Din;
        int col = el - row * Din;              // multiple of 8 (Din % 8 == 0)
        unsigned laddr = ldsbase + (unsigned)(row * LP + col) * 2u;
        const __bf16* g = Wt + el;
        asm volatile("global_load_async_to_lds_b128 %0, %1, off"
                     :: "v"(laddr), "v"(g) : "memory");
    }
    wait_asynccnt0();
    __syncthreads();
}

// --------------------------- WMMA GEMM (strip) -----------------------------
// Y[M x (NT*16)] = act( (X (+X2)) [M x Din] * W + bias ).
// Block: 256 threads = 8 waves; wave w computes rows [16*(8*bid+w), +16) for
// the whole Dout. B fragments served from LDS. Din % 32 == 0, M % 16 == 0.
template <int NT>
__global__ __launch_bounds__(256) void wmma_gemm_strip(
    const float* __restrict__ X, const float* __restrict__ X2, int addX2,
    const __bf16* __restrict__ Wt, const float* __restrict__ bias,
    float* __restrict__ Y, int M, int Din, int act)
{
    extern __shared__ char smem_raw[];
    __bf16* sW = (__bf16*)smem_raw;
    const int Dout = NT * 16;
    const int LP   = Din + 16;                 // padded pitch (32B), bank rotate

    stage_weights_async(Wt, sW, Din, Dout, LP);

    int tilesM = M >> 4;
    int tm = blockIdx.x * 8 + (threadIdx.x >> 5);
    if (tm >= tilesM) return;                  // wave-uniform
    int m0   = tm << 4;
    int lane = threadIdx.x & 31;
    int half = lane >> 4;                      // 0: lanes 0-15, 1: lanes 16-31
    int lm   = lane & 15;

    v8f acc[NT];
#pragma unroll
    for (int j = 0; j < NT; ++j) {
        float bv = bias[j * 16 + lm];
#pragma unroll
        for (int r = 0; r < 8; ++r) acc[j][r] = bv;
    }

    const float* arow  = X  + (size_t)(m0 + lm) * Din;
    const float* arow2 = X2 + (size_t)(m0 + lm) * Din;

    for (int k0 = 0; k0 < Din; k0 += 32) {
        // A fragment: lane<16 holds K = k0+{0..15},
        // lane>=16 holds K = k0+{16..31}; element order = ascending K.
        int c0 = k0 + (half ? 16 : 0);
        float af[16];
#pragma unroll
        for (int i = 0; i < 8; ++i) af[i]     = arow[c0 + i];
#pragma unroll
        for (int i = 0; i < 8; ++i) af[8 + i] = arow[c0 + 8 + i];
        if (addX2) {
#pragma unroll
            for (int i = 0; i < 8; ++i) af[i]     += arow2[c0 + i];
#pragma unroll
            for (int i = 0; i < 8; ++i) af[8 + i] += arow2[c0 + 8 + i];
        }
        v16bf a;
#pragma unroll
        for (int i = 0; i < 16; ++i) a[i] = (__bf16)af[i];

#pragma unroll
        for (int j = 0; j < NT; ++j) {
            // B fragment: lane holds N = j*16+lm, contiguous K run from LDS.
            const v16bf* bp = reinterpret_cast<const v16bf*>(
                sW + (size_t)(j * 16 + lm) * LP + half * 16 + k0);
            acc[j] = __builtin_amdgcn_wmma_f32_16x16x32_bf16(
                false, a, false, *bp, (short)0, acc[j], false, false);
        }
    }

    // C/D layout: lane lm -> column, VGPR r -> row m0 + r + half*8.
#pragma unroll
    for (int j = 0; j < NT; ++j) {
        float* ycol = Y + (size_t)m0 * Dout + j * 16 + lm;
#pragma unroll
        for (int r = 0; r < 8; ++r) {
            int rr = r + half * 8;
            ycol[(size_t)rr * Dout] = apply_act(acc[j][r], act);
        }
    }
}

// --------------------- WMMA GEMM over edge concat --------------------------
// A row e = [ x[row[e]], x[col[e]], ea[e] ], Din = 384, Dout = 128.
__global__ __launch_bounds__(256) void wmma_gemm_edgecat(
    const float* __restrict__ x, const float* __restrict__ ea,
    const int* __restrict__ rowi, const int* __restrict__ coli,
    const __bf16* __restrict__ Wt, const float* __restrict__ bias,
    float* __restrict__ Y, int E, int act)
{
    extern __shared__ char smem_raw[];
    __bf16* sW = (__bf16*)smem_raw;
    const int Din = 384, NT = 8, Dout = 128;
    const int LP  = Din + 16;

    stage_weights_async(Wt, sW, Din, Dout, LP);

    int tilesM = E >> 4;
    int tm = blockIdx.x * 8 + (threadIdx.x >> 5);
    if (tm >= tilesM) return;
    int m0   = tm << 4;
    int lane = threadIdx.x & 31;
    int half = lane >> 4;
    int lm   = lane & 15;

    int e = m0 + lm;
    const float* s0 = x  + (size_t)rowi[e] * 128;
    const float* s1 = x  + (size_t)coli[e] * 128;
    const float* s2 = ea + (size_t)e       * 128;

    v8f acc[NT];
#pragma unroll
    for (int j = 0; j < NT; ++j) {
        float bv = bias[j * 16 + lm];
#pragma unroll
        for (int r = 0; r < 8; ++r) acc[j][r] = bv;
    }

    for (int k0 = 0; k0 < Din; k0 += 32) {
        int c0 = k0 + (half ? 16 : 0);
        int c1 = c0 + 8;
        const float* p0 = (c0 < 128) ? (s0 + c0)
                        : (c0 < 256) ? (s1 + (c0 - 128)) : (s2 + (c0 - 256));
        const float* p1 = (c1 < 128) ? (s0 + c1)
                        : (c1 < 256) ? (s1 + (c1 - 128)) : (s2 + (c1 - 256));
        float af[16];
#pragma unroll
        for (int i = 0; i < 8; ++i) af[i]     = p0[i];
#pragma unroll
        for (int i = 0; i < 8; ++i) af[8 + i] = p1[i];
        v16bf a;
#pragma unroll
        for (int i = 0; i < 16; ++i) a[i] = (__bf16)af[i];

#pragma unroll
        for (int j = 0; j < NT; ++j) {
            const v16bf* bp = reinterpret_cast<const v16bf*>(
                sW + (size_t)(j * 16 + lm) * LP + half * 16 + k0);
            acc[j] = __builtin_amdgcn_wmma_f32_16x16x32_bf16(
                false, a, false, *bp, (short)0, acc[j], false, false);
        }
    }

#pragma unroll
    for (int j = 0; j < NT; ++j) {
        float* ycol = Y + (size_t)m0 * Dout + j * 16 + lm;
#pragma unroll
        for (int r = 0; r < 8; ++r) {
            int rr = r + half * 8;
            ycol[(size_t)rr * Dout] = apply_act(acc[j][r], act);
        }
    }
}

// ------------------------- weight convert/transpose ------------------------
// W f32 [DinSrc x Dout]  ->  Wt bf16 [Dout x DinPad], zero-padded K tail.
__global__ void conv_wt(const float* __restrict__ W, __bf16* __restrict__ Wt,
                        int DinSrc, int DinPad, int Dout)
{
    int t = blockIdx.x * blockDim.x + threadIdx.x;
    if (t >= DinPad * Dout) return;
    int n = t / DinPad, k = t % DinPad;
    float v = (k < DinSrc) ? W[(size_t)k * Dout + n] : 0.0f;
    Wt[(size_t)n * DinPad + k] = (__bf16)v;
}

// ------------------------------ utilities ----------------------------------
__global__ void zero_f32(float* __restrict__ p, long long n) {
    long long t = (long long)blockIdx.x * blockDim.x + threadIdx.x;
    if (t < n) p[t] = 0.0f;
}
__global__ void copy_f32(float* __restrict__ d, const float* __restrict__ s, long long n) {
    long long t = (long long)blockIdx.x * blockDim.x + threadIdx.x;
    if (t < n) d[t] = s[t];
}
__global__ void add3_f32(float* __restrict__ o, const float* __restrict__ a,
                         const float* __restrict__ b, const float* __restrict__ c,
                         long long n) {
    long long t = (long long)blockIdx.x * blockDim.x + threadIdx.x;
    if (t < n) o[t] = a[t] + b[t] + c[t];
}
// out[M x 256] = [a[M x 128] | b[M x 128]]
__global__ void concat2_f32(float* __restrict__ o, const float* __restrict__ a,
                            const float* __restrict__ b, int M) {
    long long t = (long long)blockIdx.x * blockDim.x + threadIdx.x;
    if (t >= (long long)M * 256) return;
    int i = (int)(t >> 8), c = (int)(t & 255);
    o[t] = (c < 128) ? a[(size_t)i * 128 + c] : b[(size_t)i * 128 + (c - 128)];
}

// Row-wise L2 normalize (D=128 fixed), x = x / max(||x||, 1e-12) * scale.
__global__ __launch_bounds__(128) void l2norm_rows(float* __restrict__ x, float scale) {
    __shared__ float red[4];
    size_t base = (size_t)blockIdx.x * 128;
    float v = x[base + threadIdx.x];
    float s = v * v;
#pragma unroll
    for (int m = 16; m > 0; m >>= 1) s += __shfl_xor(s, m, 32);
    if ((threadIdx.x & 31) == 0) red[threadIdx.x >> 5] = s;
    __syncthreads();
    float n = sqrtf(red[0] + red[1] + red[2] + red[3]);
    n = fmaxf(n, 1e-12f);
    x[base + threadIdx.x] = v / n * scale;
}

// ------------------------------ edge kernels -------------------------------
__global__ void edge_rbf(const float* __restrict__ Z, const int* __restrict__ rowi,
                         const int* __restrict__ coli, float* __restrict__ w,
                         float* __restrict__ rbf, int E)
{
    int e = blockIdx.x * blockDim.x + threadIdx.x;
    if (e >= E) return;
    int r = rowi[e], c = coli[e];
    float dx = Z[(size_t)r * 3 + 0] - Z[(size_t)c * 3 + 0];
    float dy = Z[(size_t)r * 3 + 1] - Z[(size_t)c * 3 + 1];
    float dz = Z[(size_t)r * 3 + 2] - Z[(size_t)c * 3 + 2];
    float d = sqrtf(dx * dx + dy * dy + dz * dz);
    w[e] = d;
    float* o = rbf + (size_t)e * 32;
    const float inv_sig = 16.0f / 6.0f;
#pragma unroll
    for (int j = 0; j < 16; ++j) {
        float mu = 6.0f * (float)j / 15.0f;
        float u = (d - mu) * inv_sig;
        o[j] = __expf(-u * u);
    }
#pragma unroll
    for (int j = 16; j < 32; ++j) o[j] = 0.0f;
}

__global__ void deg_scatter(const float* __restrict__ w, const int* __restrict__ coli,
                            float* __restrict__ deg, int E)
{
    int e = blockIdx.x * blockDim.x + threadIdx.x;
    if (e < E) atomicAdd(&deg[coli[e]], w[e]);
}
__global__ void deg_to_dinv(float* __restrict__ deg, int N) {
    int i = blockIdx.x * blockDim.x + threadIdx.x;
    if (i >= N) return;
    float d = deg[i] + 1.0f;                    // self loop weight 1
    deg[i] = (d > 0.0f) ? rsqrtf(d) : 0.0f;     // -> dinv in place
}
__global__ void edge_normw(const float* __restrict__ w, const int* __restrict__ rowi,
                           const int* __restrict__ coli, const float* __restrict__ dinv,
                           float* __restrict__ nw, int E)
{
    int e = blockIdx.x * blockDim.x + threadIdx.x;
    if (e < E) nw[e] = dinv[rowi[e]] * w[e] * dinv[coli[e]];
}

// scatter-add over edges: out[col[e]] += f(src[row[e]]), D=128.
// mode 0: src[row];  mode 1: relu(src[row] + ea[e]);  mode 2: w[e] * src[row]
__global__ void scatter_edges(const float* __restrict__ src, const float* __restrict__ ea,
                              const float* __restrict__ w, const int* __restrict__ rowi,
                              const int* __restrict__ coli, float* __restrict__ out,
                              int E, int mode)
{
    long long t = (long long)blockIdx.x * blockDim.x + threadIdx.x;
    if (t >= (long long)E * 32) return;
    int e = (int)(t >> 5);
    int c = ((int)t & 31) << 2;
    const float* s = src + (size_t)rowi[e] * 128 + c;
    float v0 = s[0], v1 = s[1], v2 = s[2], v3 = s[3];
    if (mode == 1) {
        const float* a = ea + (size_t)e * 128 + c;
        v0 = fmaxf(v0 + a[0], 0.0f); v1 = fmaxf(v1 + a[1], 0.0f);
        v2 = fmaxf(v2 + a[2], 0.0f); v3 = fmaxf(v3 + a[3], 0.0f);
    } else if (mode == 2) {
        float ww = w[e];
        v0 *= ww; v1 *= ww; v2 *= ww; v3 *= ww;
    }
    float* o = out + (size_t)coli[e] * 128 + c;
    atomicAdd(o + 0, v0); atomicAdd(o + 1, v1);
    atomicAdd(o + 2, v2); atomicAdd(o + 3, v3);
}

// segment sum of rows: out[idx[i]] += src[i], D=128
__global__ void seg_rows(const float* __restrict__ src, const int* __restrict__ idx,
                         float* __restrict__ out, int M)
{
    long long t = (long long)blockIdx.x * blockDim.x + threadIdx.x;
    if (t >= (long long)M * 32) return;
    int i = (int)(t >> 5);
    int c = ((int)t & 31) << 2;
    const float* s = src + (size_t)i * 128 + c;
    float* o = out + (size_t)idx[i] * 128 + c;
    atomicAdd(o + 0, s[0]); atomicAdd(o + 1, s[1]);
    atomicAdd(o + 2, s[2]); atomicAdd(o + 3, s[3]);
}

// x_inter = 0.9*(agg + dinv^2 * x) + 0.1*x   (self-loop folded in)
__global__ void appnp_combine(float* __restrict__ o, const float* __restrict__ agg,
                              const float* __restrict__ x, const float* __restrict__ dinv,
                              int N)
{
    long long t = (long long)blockIdx.x * blockDim.x + threadIdx.x;
    if (t >= (long long)N * 128) return;
    int i = (int)(t >> 7);
    float di = dinv[i];
    float xv = x[t];
    o[t] = 0.9f * (agg[t] + di * di * xv) + 0.1f * xv;
}

// ------------------------------ host driver --------------------------------
extern "C" void kernel_launch(void* const* d_in, const int* in_sizes, int n_in,
                              void* d_out, int out_size, void* d_ws, size_t ws_size,
                              hipStream_t stream)
{
    constexpr int N = 50000, NB = 5000, BS = 64, D = 128, E = 400000;
    (void)in_sizes; (void)n_in; (void)out_size; (void)ws_size;

    const float* H0       = (const float*)d_in[0];
    const float* Zp       = (const float*)d_in[1];
    const int*   block_id = (const int*)d_in[2];
    const int*   batch_id = (const int*)d_in[3];
    const int*   edges    = (const int*)d_in[4];
    const int*   rowi = edges;
    const int*   coli = edges + E;

    // ---- workspace bump allocator ----
    char* base = (char*)d_ws;
    size_t off = 0;
    auto alloc = [&](size_t bytes) -> void* {
        void* p = base + off;
        off = (off + bytes + 255) & ~(size_t)255;
        return p;
    };

    // ---- layers: {W idx, b idx, DinSrc, DinPad, Dout} ----
    struct Lay { int wi, bi, dinS, dinP, dout; };
    const Lay L[22] = {
        {5, 6, 128, 128, 128},    // 0  enc_lin
        {7, 8, 128, 128, 128},    // 1  lin_node
        {9, 10, 128, 128, 128},   // 2  mlp_encode
        {11, 12, 16, 32, 128},    // 3  ea_inter
        {13, 14, 384, 384, 128},  // 4  eu_inter
        {15, 16, 128, 128, 128},  // 5  gin1
        {17, 18, 128, 128, 128},  // 6  dgnn1[0]
        {19, 20, 128, 128, 128},  // 7  dgnn1[1]
        {21, 22, 128, 128, 128},  // 8  dgnn1[2]
        {23, 24, 256, 256, 128},  // 9  lin1
        {25, 26, 16, 32, 128},    // 10 ea_intra
        {27, 28, 384, 384, 128},  // 11 eu_intra
        {29, 30, 128, 128, 128},  // 12 gin3
        {31, 32, 128, 128, 128},  // 13 dgnn3[0]
        {33, 34, 128, 128, 128},  // 14 dgnn3[1]
        {35, 36, 128, 128, 128},  // 15 dgnn3[2]
        {37, 38, 256, 256, 128},  // 16 lin3
        {39, 40, 128, 128, 128},  // 17 gin4
        {41, 42, 128, 128, 256},  // 18 fc0
        {43, 44, 256, 256, 256},  // 19 fc1
        {45, 46, 256, 256, 128},  // 20 fc2
        {47, 48, 128, 128, 128},  // 21 fc3
    };
    __bf16* Wt[22];
    const float* Bias[22];
    for (int i = 0; i < 22; ++i) {
        Wt[i] = (__bf16*)alloc((size_t)L[i].dinP * L[i].dout * sizeof(__bf16));
        Bias[i] = (const float*)d_in[L[i].bi];
        int nel = L[i].dinP * L[i].dout;
        conv_wt<<<(nel + 255) / 256, 256, 0, stream>>>(
            (const float*)d_in[L[i].wi], Wt[i], L[i].dinS, L[i].dinP, L[i].dout);
    }

    // ---- node / edge buffers ----
    const size_t ND = (size_t)N * D;
    float* nb[10];
    for (int i = 0; i < 10; ++i) nb[i] = (float*)alloc(ND * 4);
    float* c0 = (float*)alloc((size_t)N * 256 * 4);   // N x 256
    float* c1 = (float*)alloc((size_t)N * 256 * 4);
    float* ew    = (float*)alloc((size_t)E * 4);
    float* nw    = (float*)alloc((size_t)E * 4);
    float* rbf   = (float*)alloc((size_t)E * 32 * 4);
    float* eaA   = (float*)alloc((size_t)E * 128 * 4);
    float* eaB   = (float*)alloc((size_t)E * 128 * 4);
    float* degv  = (float*)alloc((size_t)N * 4);      // deg -> dinv in place

    float* outH = (float*)d_out;
    float* outB = outH + ND;
    float* outG = outB + (size_t)NB * D;
    float* outZ = outG + (size_t)BS * D;

    auto gemm = [&](const float* X, const float* X2, int li, float* Y, int M, int act) {
        int grid = ((M >> 4) + 7) / 8;
        int LP = L[li].dinP + 16;
        size_t sh = (size_t)L[li].dout * LP * sizeof(__bf16);
        int addX2 = (X2 != nullptr) ? 1 : 0;
        const float* X2p = X2 ? X2 : X;
        if (L[li].dout == 256) {
            wmma_gemm_strip<16><<<grid, 256, sh, stream>>>(
                X, X2p, addX2, Wt[li], Bias[li], Y, M, L[li].dinP, act);
        } else {
            wmma_gemm_strip<8><<<grid, 256, sh, stream>>>(
                X, X2p, addX2, Wt[li], Bias[li], Y, M, L[li].dinP, act);
        }
    };
    auto gemm_ec = [&](const float* x, const float* ea, int li, float* Y, int act) {
        int grid = ((E >> 4) + 7) / 8;
        size_t sh = (size_t)128 * (384 + 16) * sizeof(__bf16);
        wmma_gemm_edgecat<<<grid, 256, sh, stream>>>(
            x, ea, rowi, coli, Wt[li], Bias[li], Y, E, act);
    };
    auto zero = [&](float* p, long long n) {
        zero_f32<<<(unsigned)((n + 255) / 256), 256, 0, stream>>>(p, n);
    };
    auto scatter = [&](const float* src, const float* ea, const float* w,
                       float* out, int mode) {
        long long tot = (long long)E * 32;
        scatter_edges<<<(unsigned)((tot + 255) / 256), 256, 0, stream>>>(
            src, ea, w, rowi, coli, out, E, mode);
    };

    // ============================ forward ============================
    edge_rbf<<<(E + 255) / 256, 256, 0, stream>>>(Zp, rowi, coli, ew, rbf, E);

    // xe = silu(enc_lin(H0)); x_psc = l2norm(xe)*1.8      -> nb0
    gemm(H0, nullptr, 0, nb[0], N, ACT_SILU);
    l2norm_rows<<<N, 128, 0, stream>>>(nb[0], 1.8f);

    // APPNP
    zero(degv, N);
    deg_scatter<<<(E + 255) / 256, 256, 0, stream>>>(ew, coli, degv, E);
    deg_to_dinv<<<(N + 255) / 256, 256, 0, stream>>>(degv, N);
    edge_normw<<<(E + 255) / 256, 256, 0, stream>>>(ew, rowi, coli, degv, nw, E);
    zero(nb[1], (long long)ND);
    scatter(nb[0], nullptr, nw, nb[1], 2);
    appnp_combine<<<(unsigned)((ND + 255) / 256), 256, 0, stream>>>(
        nb[2], nb[1], nb[0], degv, N);                 // x_inter -> nb2

    // x = bn(leaky(mlp(x_inter + silu(lin_node(H0)))))    -> nb4
    gemm(H0, nullptr, 1, nb[3], N, ACT_SILU);           // xraw -> nb3
    gemm(nb[2], nb[3], 2, nb[4], N, ACT_BN);            // x    -> nb4

    // ---- inter branch ----
    gemm(rbf, nullptr, 3, eaA, E, ACT_SIG);             // ea
    gemm_ec(nb[4], eaA, 4, eaB, ACT_SILU);              // ea = eu_inter(...)
    zero(nb[1], (long long)ND);
    scatter(nb[4], eaB, nullptr, nb[1], 1);             // agg = seg(relu(x[row]+ea))
    gemm(nb[4], nb[1], 5, nb[3], N, ACT_BN);            // xi1 -> nb3
    copy_f32<<<(unsigned)((ND + 255) / 256), 256, 0, stream>>>(nb[5], nb[4], (long long)ND);
    {
        float *h = nb[5], *h2 = nb[7], *m = nb[6];
        for (int l = 0; l < 3; ++l) {
            zero(m, (long long)ND);
            scatter(h, nullptr, nullptr, m, 0);
            gemm(h, m, 6 + l, h2, N, ACT_SILU);
            float* tmp = h; h = h2; h2 = tmp;
        }
        concat2_f32<<<(unsigned)(((long long)N * 256 + 255) / 256), 256, 0, stream>>>(
            c0, nb[3], h, N);
    }
    gemm(c0, nullptr, 9, nb[3], N, ACT_SILU);           // x_inter_b -> nb3

    // ---- intra branch ----
    gemm(rbf, nullptr, 10, eaA, E, ACT_SIG);
    gemm_ec(nb[4], eaA, 11, eaB, ACT_SILU);
    zero(nb[1], (long long)ND);
    scatter(nb[4], eaB, nullptr, nb[1], 1);
    gemm(nb[4], nb[1], 12, nb[8], N, ACT_BN);           // xa1 -> nb8
    copy_f32<<<(unsigned)((ND + 255) / 256), 256, 0, stream>>>(nb[5], nb[4], (long long)ND);
    {
        float *h = nb[5], *h2 = nb[7], *m = nb[6];
        for (int l = 0; l < 3; ++l) {
            zero(m, (long long)ND);
            scatter(h, nullptr, nullptr, m, 0);
            gemm(h, m, 13 + l, h2, N, ACT_SILU);
            float* tmp = h; h = h2; h2 = tmp;
        }
        concat2_f32<<<(unsigned)(((long long)N * 256 + 255) / 256), 256, 0, stream>>>(
            c0, nb[8], h, N);
    }
    gemm(c0, nullptr, 16, nb[8], N, ACT_SILU);          // x_intra_b -> nb8

    // x_mask = gin4                                      -> nb2
    zero(nb[1], (long long)ND);
    scatter(nb[4], nullptr, nullptr, nb[1], 0);
    gemm(nb[4], nb[1], 17, nb[2], N, ACT_BN);

    // x_c = x_inter_b + x_intra_b + x_mask               -> nb9
    add3_f32<<<(unsigned)((ND + 255) / 256), 256, 0, stream>>>(
        nb[9], nb[3], nb[8], nb[2], (long long)ND);

    // fc chain
    gemm(nb[9], nullptr, 18, c0, N, ACT_BN);            // 128 -> 256
    gemm(c0,   nullptr, 19, c1, N, ACT_BN);             // 256 -> 256
    gemm(c1,   nullptr, 20, nb[5], N, ACT_BN);          // 256 -> 128
    gemm(nb[5], nullptr, 21, outH, N, ACT_NONE);        // H_updated

    // block / graph representations
    zero(outB, (long long)NB * D);
    seg_rows<<<(unsigned)(((long long)N * 32 + 255) / 256), 256, 0, stream>>>(
        outH, block_id, outB, N);
    l2norm_rows<<<NB, 128, 0, stream>>>(outB, 1.0f);
    zero(outG, (long long)BS * D);
    seg_rows<<<(unsigned)(((long long)NB * 32 + 255) / 256), 256, 0, stream>>>(
        outB, batch_id, outG, NB);
    l2norm_rows<<<BS, 128, 0, stream>>>(outG, 1.0f);

    // Z passthrough
    copy_f32<<<(unsigned)(((long long)N * 3 + 255) / 256), 256, 0, stream>>>(
        outZ, Zp, (long long)N * 3);
}
